// QuantizedLinear_15822659518584
// MI455X (gfx1250) — matrix operand order repaired and checked by reading, not compiled, not run on hardware
//
#include <hip/hip_runtime.h>

typedef __attribute__((ext_vector_type(16))) _Float16 v16h;
typedef __attribute__((ext_vector_type(8)))  _Float16 v8h;
typedef __attribute__((ext_vector_type(8)))  float    v8f;

// Problem constants (match the reference)
constexpr int Odim = 11008, Idim = 4096, Gdim = 128;
constexpr int NG   = Idim / Gdim;     // 32 groups
constexpr int Mtot = 4 * 2048;        // 8192 rows

// Tiling: block 128x256, 8 waves of 64x64, BK=64 (two WMMA k-steps)
constexpr int BM = 128, BN = 256, BK = 64;
constexpr int LDT = BK + 8;           // 72 halfs = 144B row stride (conflict-free b128 frags)
constexpr int NSTAGE = Idim / BK;     // 64

__global__ __launch_bounds__(256)
void q4_gemm_wmma_f16(const float* __restrict__ x,
                      const int*   __restrict__ qw,
                      const float* __restrict__ scale,
                      const float* __restrict__ zp,
                      const float* __restrict__ bias,
                      float* __restrict__ out)
{
    __shared__ _Float16 As[2][BM * LDT];   // x tile [m][k] f16   (2 x 18 KB)
    __shared__ _Float16 Bs[2][BN * LDT];   // w tile [n][k] f16   (2 x 36 KB)

    const int tid = threadIdx.x;
    const int m0  = blockIdx.x * BM;
    const int n0  = blockIdx.y * BN;

    // ---- staging assignment
    // x: 128 rows -> thread tid/2 owns row, tid&1 picks 32-col half  (8 float4 / thread)
    const int xrow  = tid >> 1;
    const int xpart = tid & 1;
    const float* xg0 = x + (size_t)(m0 + xrow) * Idim + xpart * 32;
    // w: 256 rows -> thread tid owns full row slab (32 int32 = 64 nibbles / thread)
    const int* qg0 = qw + (size_t)(n0 + tid) * (Idim / 2);
    const size_t sgbase = (size_t)(n0 + tid) * NG;

    float4 xr[8];
    int4   qr[8];
    float  sc, msc;   // msc = -zp*sc  ->  w = fma(nib, sc, msc)

    auto load_stage = [&](int kk) {
        const float4* xp = (const float4*)(xg0 + kk * BK);
        #pragma unroll
        for (int j = 0; j < 8; ++j) xr[j] = xp[j];
        const int4* qp = (const int4*)(qg0 + kk * (BK / 2));
        #pragma unroll
        for (int j = 0; j < 8; ++j) qr[j] = qp[j];
        const int g = kk >> 1;                 // 64-slab inside one 128-group
        sc  = scale[sgbase + g];
        msc = -zp[sgbase + g] * sc;
    };

    auto store_stage = [&](int buf) {
        // x: f32 -> f16 via packed cvt, 8B stores
        _Float16* ap = &As[buf][xrow * LDT + xpart * 32];
        #pragma unroll
        for (int j = 0; j < 8; ++j) {
            auto p0 = __builtin_amdgcn_cvt_pkrtz(xr[j].x, xr[j].y);
            auto p1 = __builtin_amdgcn_cvt_pkrtz(xr[j].z, xr[j].w);
            uint2 st = { __builtin_bit_cast(unsigned int, p0),
                         __builtin_bit_cast(unsigned int, p1) };
            *(uint2*)(ap + j * 4) = st;
        }
        // w: dequant 2 nibbles/int with FMA, pack pairs, 16B stores
        _Float16* bp = &Bs[buf][tid * LDT];
        const int* qi = (const int*)qr;
        #pragma unroll
        for (int j = 0; j < 8; ++j) {          // 8 groups of 4 ints -> 8 halfs each
            unsigned int pw[4];
            #pragma unroll
            for (int e = 0; e < 4; ++e) {
                const int q  = qi[j * 4 + e];
                const float wl = fmaf((float)(q & 15),        sc, msc);
                const float wh = fmaf((float)((q >> 4) & 15), sc, msc);
                pw[e] = __builtin_bit_cast(unsigned int,
                            __builtin_amdgcn_cvt_pkrtz(wl, wh));
            }
            uint4 st = { pw[0], pw[1], pw[2], pw[3] };
            *(uint4*)(bp + j * 8) = st;
        }
    };

    // ---- wave tiling: 8 waves -> 2 (M) x 4 (N), wave tile 64x64
    const int lane = tid & 31;
    const int wid  = tid >> 5;
    const int wm   = wid & 1;              // 0..1 -> 64-row strip
    const int wn   = wid >> 1;             // 0..3 -> 64-col strip
    const int lr   = lane & 15;
    const int hiHalf = (lane >= 16);
    const int ca0  = hiHalf ? 8  : 0;      // A frag: split K chunks (ISA 16-bit A 16x32)
    const int cb0  = hiHalf ? 8  : 0;      // B frag: split K chunks (same as A)

    v8f acc[4][4] = {};

    auto compute_stage = [&](int buf) {
        #pragma unroll
        for (int ks = 0; ks < BK / 32; ++ks) {
            v16h af[4];
            #pragma unroll
            for (int mf = 0; mf < 4; ++mf) {
                const _Float16* p = &As[buf][(wm * 64 + mf * 16 + lr) * LDT + ks * 32 + ca0];
                v8h lo = *(const v8h*)p;
                v8h hi = *(const v8h*)(p + 16);
                af[mf] = __builtin_shufflevector(lo, hi,
                        0,1,2,3,4,5,6,7,8,9,10,11,12,13,14,15);
            }
            #pragma unroll
            for (int nf = 0; nf < 4; ++nf) {
                const _Float16* p = &Bs[buf][(wn * 64 + nf * 16 + lr) * LDT + ks * 32 + cb0];
                v8h lo = *(const v8h*)p;
                v8h hi = *(const v8h*)(p + 16);
                v16h bf = __builtin_shufflevector(lo, hi,
                        0,1,2,3,4,5,6,7,8,9,10,11,12,13,14,15);
                #pragma unroll
                for (int mf = 0; mf < 4; ++mf)
                    acc[mf][nf] = __builtin_amdgcn_wmma_f32_16x16x32_f16(
                        false, af[mf], false, bf,
                        (short)0, acc[mf][nf], false, false);
            }
        }
    };

    // ---- double-buffered main loop: one barrier per stage
    load_stage(0);
    store_stage(0);
    __syncthreads();
    for (int kk = 0; kk < NSTAGE; ++kk) {
        const int buf = kk & 1;
        if (kk + 1 < NSTAGE) {
            load_stage(kk + 1);
            if (kk + 2 < NSTAGE) {                 // global_prefetch_b8 for stage kk+2
                __builtin_prefetch(xg0 + (size_t)(kk + 2) * BK, 0, 1);
                __builtin_prefetch(qg0 + (size_t)(kk + 2) * (BK / 2), 0, 1);
            }
        }
        compute_stage(buf);
        if (kk + 1 < NSTAGE) store_stage(buf ^ 1);
        __syncthreads();
    }

    // ---- epilogue: bias + scatter per C/D f32 layout (VGPR r -> M=r or r+8)
    #pragma unroll
    for (int nf = 0; nf < 4; ++nf) {
        const int ng = n0 + wn * 64 + nf * 16 + lr;
        const float bv = bias[ng];
        #pragma unroll
        for (int mf = 0; mf < 4; ++mf) {
            const int mbase = m0 + wm * 64 + mf * 16 + (hiHalf ? 8 : 0);
            #pragma unroll
            for (int r = 0; r < 8; ++r)
                out[(size_t)(mbase + r) * Odim + ng] = acc[mf][nf][r] + bv;
        }
    }
}

extern "C" void kernel_launch(void* const* d_in, const int* in_sizes, int n_in,
                              void* d_out, int out_size, void* d_ws, size_t ws_size,
                              hipStream_t stream) {
    const float* x     = (const float*)d_in[0];
    const int*   qw    = (const int*)  d_in[1];
    const float* scale = (const float*)d_in[2];
    const float* zpv   = (const float*)d_in[3];
    const float* bias  = (const float*)d_in[4];
    float*       out   = (float*)d_out;

    dim3 grid(Mtot / BM, Odim / BN);   // 64 x 43
    q4_gemm_wmma_f16<<<grid, 256, 0, stream>>>(x, qw, scale, zpv, bias, out);
}
